// GPT2Block_25391846654179
// MI455X (gfx1250) — matrix operand order repaired and checked by reading, not compiled, not run on hardware
//
#include <hip/hip_runtime.h>
#include <math.h>

// ---------------------------------------------------------------------------
// GPT-2 block forward for MI455X (gfx1250, wave32, WMMA bf16 16x16x32,
// TDM async tile staging, permlane16 softmax reductions).
// ---------------------------------------------------------------------------

typedef __attribute__((ext_vector_type(8)))  __bf16 v8bf;
typedef __attribute__((ext_vector_type(16))) __bf16 v16bf;
typedef __attribute__((ext_vector_type(8)))  float  v8f;
typedef __attribute__((ext_vector_type(4)))  unsigned int uv4;
typedef __attribute__((ext_vector_type(4)))  int iv4;
typedef __attribute__((ext_vector_type(8)))  int iv8;

#if __has_builtin(__builtin_amdgcn_tensor_load_to_lds) && \
    __has_builtin(__builtin_amdgcn_s_wait_tensorcnt)
#define USE_TDM 1
#else
#define USE_TDM 0
#endif

#if __has_builtin(__builtin_amdgcn_permlane16)
#define USE_PERMLANE 1
#else
#define USE_PERMLANE 0
#endif

__device__ __forceinline__ v16bf join8(v8bf lo, v8bf hi) {
  return __builtin_shufflevector(lo, hi, 0,1,2,3,4,5,6,7,8,9,10,11,12,13,14,15);
}

__device__ __forceinline__ v8f wmma_bf16(v16bf a, v16bf b, v8f c) {
  // D = A(16x32 bf16) * B(32x16 bf16) + C(16x16 f32)
  return __builtin_amdgcn_wmma_f32_16x16x32_bf16(false, a, false, b,
                                                 (short)0, c, false, false);
}

// ---- cross-lane helpers (reductions within 16-lane halves) ----------------
#if USE_PERMLANE
__device__ __forceinline__ float pl16(float v, unsigned s0, unsigned s1) {
  unsigned u = __float_as_uint(v);
  unsigned r = __builtin_amdgcn_permlane16(u, u, s0, s1, false, false);
  return __uint_as_float(r);
}
__device__ __forceinline__ float halfmax16(float v) {
  v = fmaxf(v, pl16(v, 0x67452301u, 0xEFCDAB89u));  // xor 1
  v = fmaxf(v, pl16(v, 0x54761032u, 0xDCFE98BAu));  // xor 2
  v = fmaxf(v, pl16(v, 0x32107654u, 0xBA98FEDCu));  // xor 4
  v = fmaxf(v, pl16(v, 0xFEDCBA98u, 0x76543210u));  // xor 8
  return v;
}
__device__ __forceinline__ float halfsum16(float v) {
  v += pl16(v, 0x67452301u, 0xEFCDAB89u);
  v += pl16(v, 0x54761032u, 0xDCFE98BAu);
  v += pl16(v, 0x32107654u, 0xBA98FEDCu);
  v += pl16(v, 0xFEDCBA98u, 0x76543210u);
  return v;
}
#else
__device__ __forceinline__ float halfmax16(float v) {
  v = fmaxf(v, __shfl_xor(v, 1, 32));
  v = fmaxf(v, __shfl_xor(v, 2, 32));
  v = fmaxf(v, __shfl_xor(v, 4, 32));
  v = fmaxf(v, __shfl_xor(v, 8, 32));
  return v;
}
__device__ __forceinline__ float halfsum16(float v) {
  v += __shfl_xor(v, 1, 32);
  v += __shfl_xor(v, 2, 32);
  v += __shfl_xor(v, 4, 32);
  v += __shfl_xor(v, 8, 32);
  return v;
}
#endif

// ---- TDM: async 2D bf16 tile (rows x 32) global -> LDS --------------------
#if USE_TDM
__device__ __forceinline__ void tdm_load_tile32(unsigned lds_byte_addr,
                                                const __bf16* gptr,
                                                unsigned strideElems,
                                                unsigned rows) {
  unsigned long long ga = (unsigned long long)(uintptr_t)gptr;
  uv4 g0;
  g0.x = 1u;                                   // count=1, user descriptor
  g0.y = lds_byte_addr;                        // lds_addr
  g0.z = (unsigned)(ga & 0xffffffffu);         // global_addr[31:0]
  g0.w = (unsigned)((ga >> 32) & 0x01ffffffu) | (2u << 30);  // addr[56:32]|type=2
  const unsigned td0 = strideElems;            // tensor_dim0 = K
  const unsigned td1 = 1u << 20;               // tensor_dim1 (generous)
  iv8 g1;
  g1[0] = (int)(1u << 16);                     // data_size = 1 (2 bytes)
  g1[1] = (int)((td0 & 0xffffu) << 16);        // tensor_dim0[15:0]
  g1[2] = (int)((td0 >> 16) | ((td1 & 0xffffu) << 16));
  g1[3] = (int)((td1 >> 16) | (32u << 16));    // tile_dim0 = 32
  g1[4] = (int)rows;                           // tile_dim1 = rows, tile_dim2=0
  g1[5] = (int)strideElems;                    // tensor_dim0_stride[31:0]
  g1[6] = 0;                                   // stride hi / dim1_stride lo
  g1[7] = 0;
  iv4 z4 = {0, 0, 0, 0};
  iv8 z8 = {0, 0, 0, 0, 0, 0, 0, 0};
  __builtin_amdgcn_tensor_load_to_lds(g0, g1, z4, z4, z8, 0);
}
__device__ __forceinline__ unsigned lds_addr_of(const void* p) {
  // generic pointer to LDS: low 32 bits are the LDS byte offset
  return (unsigned)(uintptr_t)p;
}
#endif

// ---------------------------------------------------------------------------
// Weight convert + transpose: Wt[n*K + k] = (bf16)W[k*N + n]
// ---------------------------------------------------------------------------
__global__ __launch_bounds__(256) void wtrans_kernel(const float* __restrict__ W,
                                                     __bf16* __restrict__ Wt,
                                                     int K, int N) {
  size_t idx = (size_t)blockIdx.x * blockDim.x + threadIdx.x;
  size_t total = (size_t)K * N;
  if (idx >= total) return;
  int n = (int)(idx / (size_t)K);
  int k = (int)(idx % (size_t)K);
  Wt[idx] = (__bf16)W[(size_t)k * N + n];
}

// ---------------------------------------------------------------------------
// LayerNorm over last dim (1024), f32 in -> bf16 out. One block per row.
// ---------------------------------------------------------------------------
__global__ __launch_bounds__(256) void ln_kernel(const float* __restrict__ x,
                                                 const float* __restrict__ g,
                                                 const float* __restrict__ bta,
                                                 __bf16* __restrict__ y) {
  __shared__ float red[256];
  const int row = blockIdx.x;
  const float* xr = x + (size_t)row * 1024;
  float v[4];
  float s = 0.f;
#pragma unroll
  for (int i = 0; i < 4; ++i) { v[i] = xr[threadIdx.x + 256 * i]; s += v[i]; }
  red[threadIdx.x] = s;
  __syncthreads();
  for (int st = 128; st > 0; st >>= 1) {
    if ((int)threadIdx.x < st) red[threadIdx.x] += red[threadIdx.x + st];
    __syncthreads();
  }
  const float mu = red[0] * (1.f / 1024.f);
  __syncthreads();
  float q = 0.f;
#pragma unroll
  for (int i = 0; i < 4; ++i) { float d = v[i] - mu; q += d * d; }
  red[threadIdx.x] = q;
  __syncthreads();
  for (int st = 128; st > 0; st >>= 1) {
    if ((int)threadIdx.x < st) red[threadIdx.x] += red[threadIdx.x + st];
    __syncthreads();
  }
  const float rstd = rsqrtf(red[0] * (1.f / 1024.f) + 1e-5f);
#pragma unroll
  for (int i = 0; i < 4; ++i) {
    int c = threadIdx.x + 256 * i;
    y[(size_t)row * 1024 + c] = (__bf16)((v[i] - mu) * rstd * g[c] + bta[c]);
  }
}

// ---------------------------------------------------------------------------
// bf16 WMMA GEMM: C[M,N] = epilogue(A[M,K] @ Bt^T + bias (+resid))
//   A  : bf16 row-major [M][K]   (64x32 k-tile staged in LDS via TDM, 2 bufs)
//   Bt : bf16 transposed weight [N][K]
// 256 threads = 8 waves as 2(M) x 4(N); wave tile 32x64; block tile 64x256.
// EPI: 0 = bias -> bf16 ; 1 = bias + exact GELU -> bf16 ; 2 = bias + resid -> f32
// ---------------------------------------------------------------------------
template <int EPI>
__global__ __launch_bounds__(256) void gemm_bf16_kernel(
    const __bf16* __restrict__ A, const __bf16* __restrict__ Bt,
    const float* __restrict__ bias, const float* __restrict__ resid,
    float* __restrict__ Cf, __bf16* __restrict__ Cb, int M, int N, int K) {
  __shared__ __align__(16) __bf16 atile[2][64 * 32];

  const int lane = threadIdx.x & 31;
  const int wave = threadIdx.x >> 5;
  const int half = lane >> 4;
  const int l16  = lane & 15;
  const int wy = wave >> 2;           // 0..1 : 32-row group
  const int wx = wave & 3;            // 0..3 : 64-col group
  const int Mblk = blockIdx.y * 64;
  const int Nw   = blockIdx.x * 256 + wx * 64;
  const int asel  = half * 8;         // unused (A frags load via bksel)
  const int bksel = half * 16;        // A/B-frag K chunk (contiguous k = 16*half + s)

  const __bf16* Ablk = A + (size_t)Mblk * K;   // block's 64-row slab
  v8f acc[2][4] = {};

#if USE_TDM
  if (wave == 0) {
    tdm_load_tile32(lds_addr_of(&atile[0][0]), Ablk, (unsigned)K, 64u);
  }
#endif

  for (int k = 0; k < K; k += 32) {
    const int cur = (k >> 5) & 1;
#if USE_TDM
    if (wave == 0) __builtin_amdgcn_s_wait_tensorcnt(0);
    __syncthreads();                       // tile `cur` ready; prev reads done
    if (wave == 0 && (k + 32) < K) {
      tdm_load_tile32(lds_addr_of(&atile[cur ^ 1][0]), Ablk + k + 32,
                      (unsigned)K, 64u);
    }
#else
    // cooperative fallback: 256 threads x 16B = 4KB tile
    {
      const int row = threadIdx.x >> 2;          // 0..63
      const int ch  = (threadIdx.x & 3) * 8;     // 0,8,16,24
      *(v8bf*)(&atile[cur][row * 32 + ch]) =
          *(const v8bf*)(Ablk + (size_t)row * K + k + ch);
      __syncthreads();
    }
#endif
    const __bf16* at = &atile[cur][0];
    v16bf af[2];
#pragma unroll
    for (int f = 0; f < 2; ++f) {
      const int row = wy * 32 + f * 16 + l16;
      af[f] = join8(*(const v8bf*)(at + row * 32 + bksel),
                    *(const v8bf*)(at + row * 32 + 8 + bksel));
    }
#pragma unroll
    for (int i = 0; i < 4; ++i) {
      const __bf16* bp = Bt + (size_t)(Nw + 16 * i + l16) * K + k + bksel;
      __builtin_prefetch(bp + 32, 0, 1);   // global_prefetch_b8 (next k-step)
      v16bf bfv = *(const v16bf*)bp;
      acc[0][i] = wmma_bf16(af[0], bfv, acc[0][i]);
      acc[1][i] = wmma_bf16(af[1], bfv, acc[1][i]);
    }
#if !USE_TDM
    __syncthreads();                      // reads done before next overwrite
#endif
  }

#pragma unroll
  for (int f = 0; f < 2; ++f) {
#pragma unroll
    for (int r = 0; r < 8; ++r) {
      const int row = Mblk + wy * 32 + f * 16 + half * 8 + r;
#pragma unroll
      for (int i = 0; i < 4; ++i) {
        const int col = Nw + 16 * i + l16;
        float v = acc[f][i][r] + bias[col];
        if (EPI == 1) v = 0.5f * v * (1.f + erff(v * 0.70710678118654752f));
        if (EPI == 2) {
          v += resid[(size_t)row * N + col];
          Cf[(size_t)row * N + col] = v;
        } else {
          Cb[(size_t)row * N + col] = (__bf16)v;
        }
      }
    }
  }
}

// ---------------------------------------------------------------------------
// V transpose: vt[(b*16+h)*64 + d][t] = qkv[(b*2048+t)*3072 + 2048 + h*64 + d]
// ---------------------------------------------------------------------------
__global__ __launch_bounds__(256) void vtrans_kernel(const __bf16* __restrict__ qkv,
                                                     __bf16* __restrict__ vt) {
  size_t idx = (size_t)blockIdx.x * blockDim.x + threadIdx.x;  // B*H*64*2048
  int t = (int)(idx & 2047);
  size_t r = idx >> 11;
  int d = (int)(r & 63);  r >>= 6;
  int h = (int)(r & 15);
  int b = (int)(r >> 4);
  vt[idx] = qkv[((size_t)(b * 2048 + t)) * 3072 + 2048 + h * 64 + d];
}

// ---------------------------------------------------------------------------
// Causal flash attention, one wave per 16 query rows, 8 waves / block.
// blockIdx.x = ((b*16 + h)*16 + qblk);  qbase = qblk*128 + wave*16.
// Scores and AV both via v_wmma_f32_16x16x32_bf16.
// ---------------------------------------------------------------------------
__global__ __launch_bounds__(256) void attn_kernel(const __bf16* __restrict__ qkv,
                                                   const __bf16* __restrict__ vt,
                                                   __bf16* __restrict__ out) {
  __shared__ __align__(32) __bf16 pbuf[8][16 * 32];  // per-wave P tile

  const int lane = threadIdx.x & 31;
  const int wave = threadIdx.x >> 5;
  const int half = lane >> 4;
  const int l16  = lane & 15;
  const int asel  = half * 8;
  const int bksel = half * 16;

  int bid = blockIdx.x;
  const int qblk = bid & 15;  bid >>= 4;
  const int h    = bid & 15;  bid >>= 4;
  const int b    = bid;                       // 0..3
  const int qbase = qblk * 128 + wave * 16;   // query tile start (in T)

  // Q fragments (16 rows x 64 d = 2 K-steps of 32)
  const __bf16* qp = qkv + (size_t)(b * 2048 + qbase + l16) * 3072 + h * 64;
  v16bf qf[2];
#pragma unroll
  for (int s = 0; s < 2; ++s) {
    v8bf lo = *(const v8bf*)(qp + 32 * s + bksel);
    v8bf hi = *(const v8bf*)(qp + 32 * s + 8 + bksel);
    qf[s] = join8(lo, hi);
  }
  const __bf16* kp0 = qkv + (size_t)(b * 2048) * 3072 + 1024 + h * 64;
  const __bf16* vp0 = vt + (size_t)(b * 16 + h) * 64 * 2048;
  __bf16* pw = &pbuf[wave][0];

  v8f o[4] = {};
  float mr[8], lr[8];
#pragma unroll
  for (int r = 0; r < 8; ++r) { mr[r] = -3.0e38f; lr[r] = 0.f; }

  for (int j = 0; j < qbase + 16; j += 32) {      // 32 keys/step, causal bound
    // --- S = Q K^T (two 16x16 key tiles) ---
    v8f s0 = {}, s1 = {};
#pragma unroll
    for (int s = 0; s < 2; ++s) {
      v16bf b0 = *(const v16bf*)(kp0 + (size_t)(j + l16) * 3072 + 32 * s + bksel);
      v16bf b1 = *(const v16bf*)(kp0 + (size_t)(j + 16 + l16) * 3072 + 32 * s + bksel);
      s0 = wmma_bf16(qf[s], b0, s0);
      s1 = wmma_bf16(qf[s], b1, s1);
    }
    // --- mask + online softmax (stats per row r within 16-lane halves) ---
    const int key0 = j + l16, key1 = j + 16 + l16;
#pragma unroll
    for (int r = 0; r < 8; ++r) {
      const int qr = qbase + half * 8 + r;
      float v0 = (key0 <= qr) ? s0[r] * 0.125f : -3.0e38f;  // 1/sqrt(64)
      float v1 = (key1 <= qr) ? s1[r] * 0.125f : -3.0e38f;
      const float rowmax = halfmax16(fmaxf(v0, v1));
      const float mnew   = fmaxf(mr[r], rowmax);
      const float alpha  = __expf(mr[r] - mnew);
      const float p0 = __expf(v0 - mnew);
      const float p1 = __expf(v1 - mnew);
      const float rs = halfsum16(p0 + p1);
      lr[r] = lr[r] * alpha + rs;
      mr[r] = mnew;
#pragma unroll
      for (int t = 0; t < 4; ++t) o[t][r] *= alpha;
      const int prow = half * 8 + r;                 // C-layout row
      pw[prow * 32 + l16]      = (__bf16)p0;
      pw[prow * 32 + 16 + l16] = (__bf16)p1;
    }
    // C-layout -> A-layout via per-wave LDS tile (same-wave DS ops in-order)
    asm volatile("s_wait_dscnt 0x0" ::: "memory");
    v8bf plo = *(const v8bf*)(pw + l16 * 32 + bksel);
    v8bf phi = *(const v8bf*)(pw + l16 * 32 + 8 + bksel);
    v16bf pa = join8(plo, phi);
    // --- O += P V (K-dim = 32 keys, 4 d-tiles of 16) ---
#pragma unroll
    for (int t = 0; t < 4; ++t) {
      v16bf bv = *(const v16bf*)(vp0 + (size_t)(16 * t + l16) * 2048 + j + bksel);
      o[t] = wmma_bf16(pa, bv, o[t]);
    }
    asm volatile("" ::: "memory");
  }

  // --- normalize and write merged heads: out[row][h*64 + d] ---
#pragma unroll
  for (int r = 0; r < 8; ++r) {
    const int row = b * 2048 + qbase + half * 8 + r;
    const float inv = 1.f / lr[r];
#pragma unroll
    for (int t = 0; t < 4; ++t) {
      out[(size_t)row * 1024 + h * 64 + 16 * t + l16] = (__bf16)(o[t][r] * inv);
    }
  }
}

// ---------------------------------------------------------------------------
// Host-side orchestration
// ---------------------------------------------------------------------------
extern "C" void kernel_launch(void* const* d_in, const int* in_sizes, int n_in,
                              void* d_out, int out_size, void* d_ws, size_t ws_size,
                              hipStream_t stream) {
  (void)in_sizes; (void)n_in; (void)out_size; (void)ws_size;
  const float* x      = (const float*)d_in[0];
  const float* W_attn = (const float*)d_in[1];
  const float* b_attn = (const float*)d_in[2];
  const float* W_proj = (const float*)d_in[3];
  const float* b_proj = (const float*)d_in[4];
  const float* ln1_g  = (const float*)d_in[5];
  const float* ln1_b  = (const float*)d_in[6];
  const float* ln2_g  = (const float*)d_in[7];
  const float* ln2_b  = (const float*)d_in[8];
  const float* W_fc   = (const float*)d_in[9];
  const float* b_fc   = (const float*)d_in[10];
  const float* W_out  = (const float*)d_in[11];
  const float* b_out  = (const float*)d_in[12];
  float* outp = (float*)d_out;

  const int M = 8192;  // B*T

  char* ws = (char*)d_ws;
  size_t off = 0;
  auto take = [&](size_t bytes) -> char* {
    char* p = ws + off;
    off = (off + bytes + 255) & ~(size_t)255;
    return p;
  };
  __bf16* Wt_attn = (__bf16*)take((size_t)3072 * 1024 * 2);
  __bf16* Wt_proj = (__bf16*)take((size_t)1024 * 1024 * 2);
  __bf16* Wt_fc   = (__bf16*)take((size_t)4096 * 1024 * 2);
  __bf16* Wt_out  = (__bf16*)take((size_t)1024 * 4096 * 2);
  __bf16* ln1bf   = (__bf16*)take((size_t)M * 1024 * 2);
  __bf16* qkvbf   = (__bf16*)take((size_t)M * 3072 * 2);
  __bf16* vtbf    = (__bf16*)take((size_t)64 * 64 * 2048 * 2);
  __bf16* attbf   = (__bf16*)take((size_t)M * 1024 * 2);
  float*  x1      = (float*)take((size_t)M * 1024 * 4);
  __bf16* hbf     = (__bf16*)take((size_t)M * 4096 * 2);
  __bf16* ln2bf   = ln1bf;  // ln1 dead after QKV GEMM

  dim3 blk(256);

  // Weights -> bf16 [N][K]
  wtrans_kernel<<<(3072 * 1024 + 255) / 256, blk, 0, stream>>>(W_attn, Wt_attn, 1024, 3072);
  wtrans_kernel<<<(1024 * 1024 + 255) / 256, blk, 0, stream>>>(W_proj, Wt_proj, 1024, 1024);
  wtrans_kernel<<<(4096 * 1024 + 255) / 256, blk, 0, stream>>>(W_fc,   Wt_fc,   1024, 4096);
  wtrans_kernel<<<(4096 * 1024 + 255) / 256, blk, 0, stream>>>(W_out,  Wt_out,  4096, 1024);

  // ln1 -> QKV GEMM (bf16 out)
  ln_kernel<<<M, blk, 0, stream>>>(x, ln1_g, ln1_b, ln1bf);
  gemm_bf16_kernel<0><<<dim3(3072 / 256, M / 64), blk, 0, stream>>>(
      ln1bf, Wt_attn, b_attn, nullptr, nullptr, qkvbf, M, 3072, 1024);

  // attention
  vtrans_kernel<<<(64 * 64 * 2048) / 256, blk, 0, stream>>>(qkvbf, vtbf);
  attn_kernel<<<4 * 16 * 16, blk, 0, stream>>>(qkvbf, vtbf, attbf);

  // proj + residual -> x1 (f32)
  gemm_bf16_kernel<2><<<dim3(1024 / 256, M / 64), blk, 0, stream>>>(
      attbf, Wt_proj, b_proj, x, x1, nullptr, M, 1024, 1024);

  // ln2 -> FC (+GELU) -> out (+residual) -> d_out
  ln_kernel<<<M, blk, 0, stream>>>(x1, ln2_g, ln2_b, ln2bf);
  gemm_bf16_kernel<1><<<dim3(4096 / 256, M / 64), blk, 0, stream>>>(
      ln2bf, Wt_fc, b_fc, nullptr, nullptr, hbf, M, 4096, 1024);
  gemm_bf16_kernel<2><<<dim3(1024 / 256, M / 64), blk, 0, stream>>>(
      hbf, Wt_out, b_out, x1, outp, nullptr, M, 1024, 4096);
}
